// BOPN_Model_45380624449999
// MI455X (gfx1250) — compile-verified, matrix operand order checked by reading, not run on hardware
//
#include <hip/hip_runtime.h>

typedef __attribute__((ext_vector_type(16))) _Float16 v16h;
typedef __attribute__((ext_vector_type(8)))  _Float16 v8h;
typedef __attribute__((ext_vector_type(8)))  float    v8f;

union Ah { v16h v; v8h h[2]; _Float16 e[16]; };
union Cf { v8f  v; float    e[8];  };

#define Eb   256
#define Hh   16
#define Dd   16
#define FFd  512
#define Ll   5
#define Bb   4
#define Nn   256
#define Mrows (Bb*Nn)          // 1024
#define ME   (Bb*Nn*Eb)        // 262144

// K index for element p of a 16-bit 16x32 A/B fragment (ISA 7.12.2)
__device__ __forceinline__ constexpr int kmap(int p, int half) {
    return (p < 8) ? (half * 8 + p) : (half * 8 + p + 8);
}

// ---------------- setup kernels ----------------

__global__ void minmax_kernel(const float* __restrict__ data, float* __restrict__ mnmx) {
    int b = blockIdx.x, tid = threadIdx.x;
    float mn = 3.0e38f, mx = -3.0e38f;
    const float* d = data + b * (Nn * Nn);
    for (int i = tid; i < Nn * Nn; i += 256) {
        float v = d[i];
        mn = fminf(mn, v); mx = fmaxf(mx, v);
    }
    __shared__ float smn[256], smx[256];
    smn[tid] = mn; smx[tid] = mx;
    __syncthreads();
    for (int s = 128; s > 0; s >>= 1) {
        if (tid < s) { smn[tid] = fminf(smn[tid], smn[tid + s]); smx[tid] = fmaxf(smx[tid], smx[tid + s]); }
        __syncthreads();
    }
    if (tid == 0) { mnmx[b * 2] = smn[0]; mnmx[b * 2 + 1] = smx[0]; }
}

__global__ void scale_kernel(const float* __restrict__ data, const float* __restrict__ mnmx,
                             float* __restrict__ scaled) {
    int idx = blockIdx.x * 256 + threadIdx.x;       // < B*N*N
    int b = idx >> 16;
    float mn = mnmx[b * 2], mx = mnmx[b * 2 + 1];
    float rng = mx - mn;
    if (rng == 0.f) rng = 1.f;
    scaled[idx] = (data[idx] - mn) / rng;
}

// wem[ij,h] = sum_e Wedge[e]*Wmix[ij,e,h] ; bem analogously with bedge
__global__ void mixprep_kernel(const float* __restrict__ Wedge, const float* __restrict__ bedge,
                               const float* __restrict__ Wmix,
                               float* __restrict__ mixw, float* __restrict__ mixb) {
    int ij = blockIdx.x, h = threadIdx.x;
    if (h < Hh) {
        float sw = 0.f, sb = 0.f;
        for (int e = 0; e < Eb; e++) {
            float m = Wmix[(ij * Eb + e) * Hh + h];
            sw += Wedge[e] * m;
            sb += bedge[e] * m;
        }
        mixw[ij * Hh + h] = sw;
        mixb[ij * Hh + h] = sb;
    }
}

__global__ void embed_kernel(const float* __restrict__ nr, const float* __restrict__ Wn,
                             const float* __restrict__ bn,
                             float* __restrict__ rF, float* __restrict__ cF,
                             _Float16* __restrict__ rH, _Float16* __restrict__ cH) {
    int idx = blockIdx.x * 256 + threadIdx.x;       // < B*N*E
    int e = idx & (Eb - 1);
    int bn_ = idx >> 8;
    float v = nr[bn_] * Wn[e] + bn[e];
    rF[idx] = v; cF[idx] = v;
    rH[idx] = (_Float16)v; cH[idx] = (_Float16)v;
}

// ---------------- weight pre-swizzle ----------------
// W (f32) [nMat][K][Nc] -> Wf (f16) [nMat][K/32][Nc/16][32 lanes][16], so that a
// lane's whole 16x32 B fragment is one contiguous 32-byte run.
// grid(K*Nc/256, nMat), block 256. Writes are fully coalesced.

template <int K, int Nc>
__global__ void swizzle_kernel(const float* __restrict__ W, _Float16* __restrict__ Wf) {
    constexpr int TN = Nc / 16;
    int o = blockIdx.x * 256 + threadIdx.x;         // < K*Nc
    int mat = blockIdx.y;
    int p = o & 15;
    int lane = (o >> 4) & 31;
    int rest = o >> 9;
    int tn = rest & (TN - 1);
    int kc = rest / TN;
    int kk = kmap(p, lane >> 4);
    size_t base = (size_t)mat * K * Nc;
    Wf[base + o] = (_Float16)W[base + (kc * 32 + kk) * Nc + tn * 16 + (lane & 15)];
}

// ---------------- WMMA GEMM ----------------
// out = A(f16)[1024,K] @ W -> via pre-swizzled f16 fragments. No LDS, no barriers.
// grid(Nc/64, 4), block(32,8). Each wave: TWO 16-row strips x 4 WMMA tiles
// (8 accumulators); each B fragment is reused by 2 WMMAs.

template <int K, int Nc, bool BIAS, bool RESID, bool RELU, bool OF, bool OH>
__global__ __launch_bounds__(256)
void gemm_kernel(const _Float16* __restrict__ A, const _Float16* __restrict__ Wf,
                 const float* __restrict__ bias, const float* __restrict__ resid,
                 float* __restrict__ outF, _Float16* __restrict__ outH) {
    int lane = threadIdx.x;
    int half = lane >> 4, lc = lane & 15;
    int s0 = blockIdx.y * 16 + threadIdx.y;         // strip pair: s0, s0+8
    int r0a = s0 * 16;
    int r0b = (s0 + 8) * 16;
    int t0 = blockIdx.x * 4;                        // base n-tile index

    const _Float16* apa = A + (r0a + lc) * K + half * 8;
    const _Float16* apb = A + (r0b + lc) * K + half * 8;
    const _Float16* bp  = Wf + t0 * 512 + lane * 16;

    Cf acc0[4] = {{}, {}, {}, {}};
    Cf acc1[4] = {{}, {}, {}, {}};

#pragma unroll
    for (int kc = 0; kc < K / 32; kc++) {
        Ah a0, a1;
        a0.h[0] = *(const v8h*)(apa + kc * 32);
        a0.h[1] = *(const v8h*)(apa + kc * 32 + 16);
        a1.h[0] = *(const v8h*)(apb + kc * 32);
        a1.h[1] = *(const v8h*)(apb + kc * 32 + 16);
#pragma unroll
        for (int t = 0; t < 4; t++) {
            Ah b;
            b.v = *(const v16h*)(bp + (kc * (Nc / 16) + t) * 512);
            acc0[t].v = __builtin_amdgcn_wmma_f32_16x16x32_f16(false, a0.v, false, b.v,
                                                               (short)0, acc0[t].v, false, false);
            acc1[t].v = __builtin_amdgcn_wmma_f32_16x16x32_f16(false, a1.v, false, b.v,
                                                               (short)0, acc1[t].v, false, false);
        }
    }

    int rowb = half * 8;
#pragma unroll
    for (int t = 0; t < 4; t++) {
        int col = (t0 + t) * 16 + lc;
        float bc = BIAS ? bias[col] : 0.f;
#pragma unroll
        for (int r = 0; r < 8; r++) {
            {
                int off = (r0a + rowb + r) * Nc + col;
                float v = acc0[t].e[r] + bc;
                if (RESID) v += resid[off];
                if (RELU)  v = fmaxf(v, 0.f);
                if (OF) outF[off] = v;
                if (OH) outH[off] = (_Float16)v;
            }
            {
                int off = (r0b + rowb + r) * Nc + col;
                float v = acc1[t].e[r] + bc;
                if (RESID) v += resid[off];
                if (RELU)  v = fmaxf(v, 0.f);
                if (OF) outF[off] = v;
                if (OH) outH[off] = (_Float16)v;
            }
        }
    }
}

// ---------------- attention: per (b,h,n-tile) : softmax(QK^T/4 + bias) V ---------------
// grid(16, H, B), block(32) : one wave. Q,K,V,O are [B*N, H*D] f16.

template <int TFLAG>
__global__ __launch_bounds__(32)
void attn_kernel(const _Float16* __restrict__ Q, const _Float16* __restrict__ K,
                 const _Float16* __restrict__ V, const float* __restrict__ scaled,
                 const float* __restrict__ wem, const float* __restrict__ bem,
                 _Float16* __restrict__ O) {
    __shared__ float    S[16][Nn];
    __shared__ _Float16 Wt[16][Nn];
    __shared__ float    rinv[16];

    int lane = threadIdx.x;
    int half = lane >> 4, lc = lane & 15;
    int n0 = blockIdx.x * 16, h = blockIdx.y, bb = blockIdx.z;
    int base = bb * (Nn * 256);                     // row stride H*D = 256
    int rowb = half * 8;

    // Q fragment: d = 0..15 in one contiguous 16B run; K-dim padded 16->32 with zeros
    Ah a;
    a.h[0] = *(const v8h*)(Q + base + (n0 + lc) * 256 + h * Dd + half * 8);
#pragma unroll
    for (int p = 8; p < 16; p++) a.e[p] = (_Float16)0.f;

    // scores: 16 tiles across m
#pragma unroll 4
    for (int mt = 0; mt < 16; mt++) {
        int m0 = mt * 16;
        Ah bf;
        bf.h[0] = *(const v8h*)(K + base + (m0 + lc) * 256 + h * Dd + half * 8);
#pragma unroll
        for (int p = 8; p < 16; p++) bf.e[p] = (_Float16)0.f;
        Cf c = {};
        c.v = __builtin_amdgcn_wmma_f32_16x16x32_f16(false, a.v, false, bf.v,
                                                     (short)0, c.v, false, false);
#pragma unroll
        for (int r = 0; r < 8; r++) S[rowb + r][m0 + lc] = c.e[r];
    }
    __syncthreads();

    // softmax rows (lanes 0..15, one q-row each), fused rank-1 edge bias
    if (lane < 16) {
        int row = lane, gq = n0 + row;
        float wh = wem[h], bh = bem[h];
        const float* sp = TFLAG ? scaled + bb * (Nn * Nn) + gq
                                : scaled + bb * (Nn * Nn) + gq * Nn;
        float mx = -3.0e38f;
        for (int m = 0; m < Nn; m++) {
            float sc = TFLAG ? sp[m * Nn] : sp[m];
            float s = S[row][m] * 0.25f + sc * wh + bh;
            S[row][m] = s;
            mx = fmaxf(mx, s);
        }
        float sum = 0.f;
        for (int m = 0; m < Nn; m++) {
            float ex = __expf(S[row][m] - mx);
            sum += ex;
            Wt[row][m] = (_Float16)ex;
        }
        rinv[row] = 1.0f / sum;
    }
    __syncthreads();

    // out = P @ V : chain of 8 WMMAs over m. P fragment = two contiguous LDS b128 reads.
    const _Float16* wp = &Wt[lc][0];
    const _Float16* vp = V + base + h * Dd + lc;
    Cf acc = {};
#pragma unroll
    for (int kc = 0; kc < 8; kc++) {
        Ah a2, b2;
        a2.h[0] = *(const v8h*)(wp + kc * 32 + half * 8);
        a2.h[1] = *(const v8h*)(wp + kc * 32 + half * 8 + 16);
#pragma unroll
        for (int p = 0; p < 16; p++) {
            b2.e[p] = vp[(kc * 32 + kmap(p, half)) * 256];
        }
        acc.v = __builtin_amdgcn_wmma_f32_16x16x32_f16(false, a2.v, false, b2.v,
                                                       (short)0, acc.v, false, false);
    }
#pragma unroll
    for (int r = 0; r < 8; r++) {
        O[base + (n0 + rowb + r) * 256 + h * Dd + lc] =
            (_Float16)(acc.e[r] * rinv[rowb + r]);
    }
}

// ---------------- instance norm over node dim N per (b,e) -----------------
// grid(B*E), block(256): thread n handles x[b,n,e].

__global__ void inorm_kernel(const float* __restrict__ x, const float* __restrict__ w,
                             const float* __restrict__ b,
                             float* __restrict__ oF, _Float16* __restrict__ oH) {
    int col = blockIdx.x;
    int bb = col >> 8, e = col & (Eb - 1);
    int n = threadIdx.x;
    int off = (bb * Nn + n) * Eb + e;
    float v = x[off];
    __shared__ float red[256];
    red[n] = v;
    __syncthreads();
    for (int s = 128; s > 0; s >>= 1) { if (n < s) red[n] += red[n + s]; __syncthreads(); }
    float mean = red[0] * (1.0f / Nn);
    __syncthreads();
    float d = v - mean;
    red[n] = d * d;
    __syncthreads();
    for (int s = 128; s > 0; s >>= 1) { if (n < s) red[n] += red[n + s]; __syncthreads(); }
    float var = red[0] * (1.0f / Nn);
    float o = d * rsqrtf(var + 1e-5f) * w[e] + b[e];
    oF[off] = o;
    oH[off] = (_Float16)o;
}

// ---------------- host orchestration ----------------

struct Weights {
    const float *bcomb, *n1w, *n1b, *b1, *b2, *n2w, *n2b;
    const _Float16 *WqS, *WkS, *WvS, *WcS, *W1S, *W2S;   // swizzled f16
};

static void run_block(hipStream_t stream, const Weights& wt, int i, int j,
                      const float* x1F, const _Float16* x1H,
                      const _Float16* x2H,
                      const float* scaled, const float* mixw, const float* mixb, int tflag,
                      _Float16* qH, _Float16* kH, _Float16* vH, _Float16* oH,
                      float* tmpF, float* o1F, _Float16* o1H, _Float16* ffH,
                      float* outF, _Float16* outH) {
    int ij = i * 2 + j;
    const _Float16* Wq_ = wt.WqS + (size_t)ij * Eb * 256;
    const _Float16* Wk_ = wt.WkS + (size_t)ij * Eb * 256;
    const _Float16* Wv_ = wt.WvS + (size_t)ij * Eb * 256;
    const _Float16* Wc_ = wt.WcS + (size_t)ij * 256 * Eb;
    const _Float16* W1_ = wt.W1S + (size_t)ij * Eb * FFd;
    const _Float16* W2_ = wt.W2S + (size_t)ij * FFd * Eb;
    const float* bc_ = wt.bcomb + ij * Eb;
    const float* b1_ = wt.b1 + ij * FFd;
    const float* b2_ = wt.b2 + ij * Eb;

    dim3 blk(32, 8);
    dim3 g256(Eb / 64, 4);     // Nc=256
    dim3 g512(FFd / 64, 4);    // Nc=512

    // QKV (no bias) -> f16
    gemm_kernel<256, 256, false, false, false, false, true>
        <<<g256, blk, 0, stream>>>(x1H, Wq_, nullptr, nullptr, nullptr, qH);
    gemm_kernel<256, 256, false, false, false, false, true>
        <<<g256, blk, 0, stream>>>(x2H, Wk_, nullptr, nullptr, nullptr, kH);
    gemm_kernel<256, 256, false, false, false, false, true>
        <<<g256, blk, 0, stream>>>(x2H, Wv_, nullptr, nullptr, nullptr, vH);

    if (tflag)
        attn_kernel<1><<<dim3(16, Hh, Bb), 32, 0, stream>>>(qH, kH, vH, scaled,
                                                            mixw + ij * Hh, mixb + ij * Hh, oH);
    else
        attn_kernel<0><<<dim3(16, Hh, Bb), 32, 0, stream>>>(qH, kH, vH, scaled,
                                                            mixw + ij * Hh, mixb + ij * Hh, oH);

    // comb + bias + residual -> f32
    gemm_kernel<256, 256, true, true, false, true, false>
        <<<g256, blk, 0, stream>>>(oH, Wc_, bc_, x1F, tmpF, nullptr);
    inorm_kernel<<<Bb * Eb, 256, 0, stream>>>(tmpF, wt.n1w + ij * Eb,
                                              wt.n1b + ij * Eb, o1F, o1H);

    // FF1 + bias + relu -> f16
    gemm_kernel<256, 512, true, false, true, false, true>
        <<<g512, blk, 0, stream>>>(o1H, W1_, b1_, nullptr, nullptr, ffH);
    // FF2 + bias + residual -> f32
    gemm_kernel<512, 256, true, true, false, true, false>
        <<<g256, blk, 0, stream>>>(ffH, W2_, b2_, o1F, tmpF, nullptr);
    inorm_kernel<<<Bb * Eb, 256, 0, stream>>>(tmpF, wt.n2w + ij * Eb,
                                              wt.n2b + ij * Eb, outF, outH);
}

extern "C" void kernel_launch(void* const* d_in, const int* in_sizes, int n_in,
                              void* d_out, int out_size, void* d_ws, size_t ws_size,
                              hipStream_t stream) {
    (void)in_sizes; (void)n_in; (void)out_size; (void)ws_size;

    const float* data      = (const float*)d_in[0];
    const float* node_rand = (const float*)d_in[1];
    const float* Wnode     = (const float*)d_in[2];
    const float* bnode     = (const float*)d_in[3];
    const float* Wedge     = (const float*)d_in[4];
    const float* bedge     = (const float*)d_in[5];
    const float* Wq    = (const float*)d_in[6];
    const float* Wk    = (const float*)d_in[7];
    const float* Wv    = (const float*)d_in[8];
    const float* Wcomb = (const float*)d_in[9];
    const float* W1    = (const float*)d_in[13];
    const float* W2    = (const float*)d_in[15];
    const float* Wmix  = (const float*)d_in[19];

    Weights wt;
    wt.bcomb = (const float*)d_in[10];
    wt.n1w   = (const float*)d_in[11];
    wt.n1b   = (const float*)d_in[12];
    wt.b1    = (const float*)d_in[14];
    wt.b2    = (const float*)d_in[16];
    wt.n2w   = (const float*)d_in[17];
    wt.n2b   = (const float*)d_in[18];

    char* p = (char*)d_ws;
    auto alloc = [&](size_t bytes) -> void* {
        void* r = (void*)p;
        p += (bytes + 255) & ~(size_t)255;
        return r;
    };

    float* mnmx   = (float*)alloc(2 * Bb * sizeof(float));
    float* scaled = (float*)alloc((size_t)Bb * Nn * Nn * sizeof(float));
    float* mixw   = (float*)alloc(Ll * 2 * Hh * sizeof(float));
    float* mixb   = (float*)alloc(Ll * 2 * Hh * sizeof(float));

    // swizzled f16 weights
    const size_t WSZ = (size_t)Ll * 2 * Eb * 256;       // 655360 elems
    const size_t WSZF = (size_t)Ll * 2 * Eb * FFd;      // 1310720 elems
    _Float16* WqS = (_Float16*)alloc(WSZ * 2);
    _Float16* WkS = (_Float16*)alloc(WSZ * 2);
    _Float16* WvS = (_Float16*)alloc(WSZ * 2);
    _Float16* WcS = (_Float16*)alloc(WSZ * 2);
    _Float16* W1S = (_Float16*)alloc(WSZF * 2);
    _Float16* W2S = (_Float16*)alloc(WSZF * 2);
    wt.WqS = WqS; wt.WkS = WkS; wt.WvS = WvS; wt.WcS = WcS; wt.W1S = W1S; wt.W2S = W2S;

    float*    sF[4];
    _Float16* sH[4];
    for (int i = 0; i < 4; i++) {
        sF[i] = (float*)alloc((size_t)ME * sizeof(float));
        sH[i] = (_Float16*)alloc((size_t)ME * sizeof(_Float16));
    }
    float*    o1F  = (float*)alloc((size_t)ME * sizeof(float));
    _Float16* o1H  = (_Float16*)alloc((size_t)ME * sizeof(_Float16));
    float*    tmpF = (float*)alloc((size_t)ME * sizeof(float));
    _Float16* qH   = (_Float16*)alloc((size_t)ME * sizeof(_Float16));
    _Float16* kH   = (_Float16*)alloc((size_t)ME * sizeof(_Float16));
    _Float16* vH   = (_Float16*)alloc((size_t)ME * sizeof(_Float16));
    _Float16* oH   = (_Float16*)alloc((size_t)ME * sizeof(_Float16));
    _Float16* ffH  = (_Float16*)alloc((size_t)Mrows * FFd * sizeof(_Float16));

    // one-time weight swizzle (f32 -> fragment-layout f16)
    swizzle_kernel<256, 256><<<dim3(Eb * 256 / 256, Ll * 2), 256, 0, stream>>>(Wq, WqS);
    swizzle_kernel<256, 256><<<dim3(Eb * 256 / 256, Ll * 2), 256, 0, stream>>>(Wk, WkS);
    swizzle_kernel<256, 256><<<dim3(Eb * 256 / 256, Ll * 2), 256, 0, stream>>>(Wv, WvS);
    swizzle_kernel<256, 256><<<dim3(Eb * 256 / 256, Ll * 2), 256, 0, stream>>>(Wcomb, WcS);
    swizzle_kernel<256, 512><<<dim3(Eb * FFd / 256, Ll * 2), 256, 0, stream>>>(W1, W1S);
    swizzle_kernel<512, 256><<<dim3(FFd * Eb / 256, Ll * 2), 256, 0, stream>>>(W2, W2S);

    minmax_kernel<<<Bb, 256, 0, stream>>>(data, mnmx);
    scale_kernel<<<(Bb * Nn * Nn) / 256, 256, 0, stream>>>(data, mnmx, scaled);
    mixprep_kernel<<<Ll * 2, 32, 0, stream>>>(Wedge, bedge, Wmix, mixw, mixb);
    embed_kernel<<<ME / 256, 256, 0, stream>>>(node_rand, Wnode, bnode,
                                               sF[0], sF[1], sH[0], sH[1]);

    float*    rF = sF[0]; _Float16* rH = sH[0];
    float*    cF = sF[1]; _Float16* cH = sH[1];
    float*    rF2 = sF[2]; _Float16* rH2 = sH[2];
    float*    cF2 = sF[3]; _Float16* cH2 = sH[3];

    for (int i = 0; i < Ll; i++) {
        // row_new = block(row, col, edge)            (j=0, edge as-is)
        run_block(stream, wt, i, 0, rF, rH, cH, scaled, mixw, mixb, 0,
                  qH, kH, vH, oH, tmpF, o1F, o1H, ffH, rF2, rH2);
        // col_new = block(col, row_old, edge^T)      (j=1, transposed edge)
        run_block(stream, wt, i, 1, cF, cH, rH, scaled, mixw, mixb, 1,
                  qH, kH, vH, oH, tmpF, o1F, o1H, ffH, cF2, cH2);
        float* tf; _Float16* th;
        tf = rF; rF = rF2; rF2 = tf;  th = rH; rH = rH2; rH2 = th;
        tf = cF; cF = cF2; cF2 = tf;  th = cH; cH = cH2; cH2 = th;
    }

    hipMemcpyAsync(d_out, rF, (size_t)ME * sizeof(float),
                   hipMemcpyDeviceToDevice, stream);
    hipMemcpyAsync((float*)d_out + ME, cF, (size_t)ME * sizeof(float),
                   hipMemcpyDeviceToDevice, stream);
}
